// PAINN_2_5987184410902
// MI455X (gfx1250) — matrix operand order repaired and checked by reading, not compiled, not run on hardware
//
#include <hip/hip_runtime.h>
#include <hip/hip_bf16.h>
#include <cstdint>
#include <cstddef>

#define N_NODES   25000
#define N_EDGES   250000
#define N_GRAPHS  512
#define DIM       128

typedef __bf16 bf16;
typedef __attribute__((ext_vector_type(16))) __bf16 v16bf;
typedef __attribute__((ext_vector_type(2)))  __bf16 v2bf;
typedef __attribute__((ext_vector_type(8)))  float  v8f;

static __device__ __forceinline__ v8f wmma_bf16(v16bf a, v16bf b, v8f c) {
  // D = A(16x32 bf16) x B(32x16 bf16) + C(16x16 f32)
  return __builtin_amdgcn_wmma_f32_16x16x32_bf16(false, a, false, b, (short)0, c, false, false);
}

// fast silu: x * v_rcp(1 + exp(-x))  (no IEEE div chains)
static __device__ __forceinline__ float silu_f(float x) {
  return x * __builtin_amdgcn_rcpf(1.0f + __expf(-x));
}

// ---- fragment builders (ISA 7.12.2 layouts, wave32) ----------------------

// A: 16x32 bf16 tile starting at column k0 of a 16x128 bf16 LDS tile.
// Element pairs (2v,2v+1) are k-consecutive -> 8x ds_load_b32 per fragment.
static __device__ __forceinline__ v16bf fragA_lds(const bf16* tile, int k0, int lane) {
  const int m = lane & 15, half = lane >> 4;
  union { v16bf v; unsigned u[8]; } r;
#pragma unroll
  for (int v = 0; v < 8; ++v) {
    const int k = ((v >> 2) << 4) + (half << 3) + ((v & 3) << 1);
    r.u[v] = *(const unsigned*)(tile + m * DIM + k0 + k);
  }
  return r.v;
}

// A built from a 16x32 f32 LDS tile (RBF values, K padded to 32 with zeros)
static __device__ __forceinline__ v16bf fragA_f32_32(const float* tile, int lane) {
  const int m = lane & 15, half = lane >> 4;
  v16bf a;
#pragma unroll
  for (int e = 0; e < 16; ++e) {
    const int v = e >> 1, p = e & 1;
    const int k = ((v >> 2) << 4) + (half << 3) + ((v & 3) << 1) + p;
    a[e] = (bf16)tile[m * 32 + k];
  }
  return a;
}

// B fragment from weights PRE-PACKED into per-lane layout:
// Wp[((kt*ntiles + nt)*32 + lane)*16 + e]  -> one 32B contiguous load per lane.
static __device__ __forceinline__ v16bf fragB_p(const bf16* __restrict__ Wp, int ntiles,
                                                int kt, int nt, int lane) {
  return *(const v16bf*)(Wp + (((size_t)kt * ntiles + nt) * 32 + lane) * 16);
}

// stage one 16x128 f32 row-tile (per-row base pointers) into bf16 LDS
static __device__ __forceinline__ void stage_row_bf16(bf16* dst_row, const float* src_row) {
  const float4* s4 = (const float4*)src_row;
#pragma unroll
  for (int q = 0; q < 16; ++q) {
    float4 f = s4[q];
    v2bf t0; t0[0] = (bf16)f.x; t0[1] = (bf16)f.y;
    v2bf t1; t1[0] = (bf16)f.z; t1[1] = (bf16)f.w;
    *(v2bf*)(dst_row + q * 4)     = t0;
    *(v2bf*)(dst_row + q * 4 + 2) = t1;
  }
}

// ---- setup kernels -------------------------------------------------------

// Pack a KxN f32 row-major weight into per-lane WMMA B-fragment order (bf16).
// K multiple of 32 (Ksrc rows valid, rest zero), N multiple of 16.
__global__ void pack_B_kernel(const float* __restrict__ W, bf16* __restrict__ dst,
                              int K, int Ksrc, int N) {
  int idx = blockIdx.x * 256 + threadIdx.x;
  if (idx >= K * N) return;
  const int e    = idx & 15;
  const int lane = (idx >> 4) & 31;
  const int tile = idx >> 9;
  const int ntiles = N >> 4;
  const int nt = tile % ntiles;
  const int kt = tile / ntiles;
  const int k = kt * 32 + ((e >> 3) << 4) + ((lane >> 4) << 3) + (e & 7);
  const int n = nt * 16 + (lane & 15);
  dst[idx] = (k < Ksrc) ? (bf16)W[(size_t)k * N + n] : (bf16)0.0f;
}

__global__ void gather_embed_kernel(const float* __restrict__ emb, const int* __restrict__ types,
                                    float* __restrict__ s0, float* __restrict__ smsg) {
  int node = blockIdx.x, c = threadIdx.x;
  float v = emb[(size_t)types[node] * DIM + c];
  s0[(size_t)node * DIM + c]   = v;
  smsg[(size_t)node * DIM + c] = v;
}

// ---- edge message kernel (fused RBF-filter GEMM + phi MLP + scatter) -----
// One wave per 16-edge tile. N_EDGES = 15625*16 exactly.
__global__ __launch_bounds__(32) void edge_msg_kernel(
    const float* __restrict__ edge_vec,
    const float* __restrict__ state0,
    const bf16* __restrict__ phiW1p, const float* __restrict__ phi_b1,
    const bf16* __restrict__ phiW2p, const float* __restrict__ phi_b2,
    const bf16* __restrict__ filtWp, const float* __restrict__ filt_b,
    const int* __restrict__ node_from, const int* __restrict__ node_to,
    float* __restrict__ state_msg, float* __restrict__ state_vec) {
  __shared__ __align__(16) bf16  inp[16 * DIM];
  __shared__ __align__(16) bf16  Hld[16 * DIM];
  __shared__ __align__(16) float rbf[16 * 32];
  __shared__ float fcv[16];
  __shared__ float nrm[16 * 3];
  __shared__ int   sfrom[16];
  __shared__ int   sto[16];

  const int lane = threadIdx.x;
  const int half = lane >> 4, nlo = lane & 15;
  const int e0 = blockIdx.x * 16;

  if (lane < 16) {
    const int e = e0 + lane;
    const float ex = edge_vec[e * 3 + 0], ey = edge_vec[e * 3 + 1], ez = edge_vec[e * 3 + 2];
    const float d = sqrtf(ex * ex + ey * ey + ez * ez);
    const float inv = __builtin_amdgcn_rcpf(d);
    nrm[lane * 3 + 0] = ex * inv; nrm[lane * 3 + 1] = ey * inv; nrm[lane * 3 + 2] = ez * inv;
    const float x = 3.14159265358979323846f * d * 0.1f;     // pi*d/R_CUT
    fcv[lane] = (d < 10.0f) ? 0.5f * (__cosf(x) + 1.0f) : 0.0f;
#pragma unroll
    for (int k = 0; k < 32; ++k)
      rbf[lane * 32 + k] = (k < 20) ? __sinf((float)(k + 1) * x) * inv : 0.0f;
    sfrom[lane] = node_from[e];
    sto[lane]   = node_to[e];
  }
  __syncthreads();

  // stage gathered node states as bf16 (each lane: one half-row, float4 loads)
  {
    const int r = lane >> 1, c0 = (lane & 1) * 64;
    stage_row_bf16(inp + r * DIM + c0, state0 + (size_t)sfrom[r] * DIM + c0);
  }
  __syncthreads();

  // GEMM1: H = silu(inp @ phi_W1 + b1)  (16x128)
  for (int nt = 0; nt < 8; ++nt) {
    const int n0 = nt * 16;
    v8f acc; const float b = phi_b1[n0 + nlo];
#pragma unroll
    for (int r = 0; r < 8; ++r) acc[r] = b;
    for (int kt = 0; kt < 4; ++kt)
      acc = wmma_bf16(fragA_lds(inp, kt * 32, lane), fragB_p(phiW1p, 8, kt, nt, lane), acc);
#pragma unroll
    for (int r = 0; r < 8; ++r)
      Hld[(r + 8 * half) * DIM + n0 + nlo] = (bf16)silu_f(acc[r]);
  }
  __syncthreads();

  const v16bf arbf = fragA_f32_32(rbf, lane);   // RBF as a K=32 A fragment

  // GEMM2 (live 2/3): phi cols 128..383; W_out fused via one WMMA per n-tile.
  // NUM_ROUNDS==1 and state_vec starts at zero -> m1 branch is dead.
  for (int nt = 8; nt < 24; ++nt) {
    const int n0 = nt * 16;
    v8f acc; const float b = phi_b2[n0 + nlo];
#pragma unroll
    for (int r = 0; r < 8; ++r) acc[r] = b;
    for (int kt = 0; kt < 4; ++kt)
      acc = wmma_bf16(fragA_lds(Hld, kt * 32, lane), fragB_p(phiW2p, 24, kt, nt, lane), acc);
    v8f wacc; const float fb = filt_b[n0 + nlo];
#pragma unroll
    for (int r = 0; r < 8; ++r) wacc[r] = fb;
    wacc = wmma_bf16(arbf, fragB_p(filtWp, 24, 0, nt, lane), wacc);
#pragma unroll
    for (int r = 0; r < 8; ++r) {
      const int m = r + 8 * half;
      const float pf = acc[r] * (wacc[r] * fcv[m]);
      const int to = sto[m];
      if (nt < 16) {                         // m2 -> state scatter
        atomicAdd(state_msg + (size_t)to * DIM + (n0 - 128 + nlo), pf);
      } else {                               // m3 -> normalized x m3 vec scatter
        const int n = n0 - 256 + nlo;
        float* dst = state_vec + (size_t)to * 384 + n;
        atomicAdd(dst,       nrm[m * 3 + 0] * pf);
        atomicAdd(dst + 128, nrm[m * 3 + 1] * pf);
        atomicAdd(dst + 256, nrm[m * 3 + 2] * pf);
      }
    }
  }
}

// ---- node update kernel (Up/Vp, UV, Vn, update MLP, graph pooling) -------
__global__ __launch_bounds__(32) void node_update_kernel(
    float* __restrict__ state,              // in/out (post-message state)
    const float* __restrict__ state_vec,
    const bf16* __restrict__ UWp, const bf16* __restrict__ VWp,
    const bf16* __restrict__ updW1p, const float* __restrict__ upd_b1,
    const bf16* __restrict__ updW2p, const float* __restrict__ upd_b2,
    const int* __restrict__ node_graph_index,
    float* __restrict__ graph_acc) {
  __shared__ __align__(16) bf16  sv3[3 * 16 * DIM];
  __shared__ __align__(16) bf16  stA[16 * DIM];
  __shared__ __align__(16) bf16  VnA[16 * DIM];
  __shared__ __align__(16) bf16  H2[16 * DIM];
  __shared__ float UVl[16 * DIM];
  __shared__ float asv[16 * DIM];

  const int lane = threadIdx.x;
  const int half = lane >> 4, nlo = lane & 15;
  const int row0 = blockIdx.x * 16;

  {
    const int r = lane >> 1, c0 = (lane & 1) * 64;
    int node = row0 + r; if (node >= N_NODES) node = N_NODES - 1;
    stage_row_bf16(stA + r * DIM + c0, state + (size_t)node * DIM + c0);
    for (int a = 0; a < 3; ++a)
      stage_row_bf16(sv3 + a * 16 * DIM + r * DIM + c0,
                     state_vec + (size_t)node * 384 + a * DIM + c0);
  }
  __syncthreads();

  // UV = sum_a Up_a*Vp_a ; Vn = ||Vp|| over axis
  for (int nt = 0; nt < 8; ++nt) {
    float uv[8], vn2[8];
#pragma unroll
    for (int r = 0; r < 8; ++r) { uv[r] = 0.f; vn2[r] = 0.f; }
    for (int a = 0; a < 3; ++a) {
      v8f up = {}, vp = {};
      for (int kt = 0; kt < 4; ++kt) {
        const v16bf af = fragA_lds(sv3 + a * 16 * DIM, kt * 32, lane);
        up = wmma_bf16(af, fragB_p(UWp, 8, kt, nt, lane), up);
        vp = wmma_bf16(af, fragB_p(VWp, 8, kt, nt, lane), vp);
      }
#pragma unroll
      for (int r = 0; r < 8; ++r) { uv[r] += up[r] * vp[r]; vn2[r] += vp[r] * vp[r]; }
    }
#pragma unroll
    for (int r = 0; r < 8; ++r) {
      const int m = r + 8 * half, n = nt * 16 + nlo;
      UVl[m * DIM + n] = uv[r];
      VnA[m * DIM + n] = (bf16)sqrtf(vn2[r]);
    }
  }
  __syncthreads();

  // H2 = silu([Vn, state] @ upd_W1 + b1)   (K = 256)
  for (int nt = 0; nt < 8; ++nt) {
    v8f acc; const float b = upd_b1[nt * 16 + nlo];
#pragma unroll
    for (int r = 0; r < 8; ++r) acc[r] = b;
    for (int kt = 0; kt < 8; ++kt) {
      const v16bf af = (kt < 4) ? fragA_lds(VnA, kt * 32, lane)
                                : fragA_lds(stA, (kt - 4) * 32, lane);
      acc = wmma_bf16(af, fragB_p(updW1p, 8, kt, nt, lane), acc);
    }
#pragma unroll
    for (int r = 0; r < 8; ++r)
      H2[(r + 8 * half) * DIM + nt * 16 + nlo] = (bf16)silu_f(acc[r]);
  }
  __syncthreads();

  const int g0 = node_graph_index[0];

  // a = H2 @ upd_W2 + b2 ; a_vv (cols 0..127) is dead (state_vec unused afterwards)
  for (int nt = 8; nt < 24; ++nt) {
    const int n0 = nt * 16;
    v8f acc; const float b = upd_b2[n0 + nlo];
#pragma unroll
    for (int r = 0; r < 8; ++r) acc[r] = b;
    for (int kt = 0; kt < 4; ++kt)
      acc = wmma_bf16(fragA_lds(H2, kt * 32, lane), fragB_p(updW2p, 24, kt, nt, lane), acc);
    if (nt < 16) {                                   // a_sv
#pragma unroll
      for (int r = 0; r < 8; ++r)
        asv[(r + 8 * half) * DIM + (n0 - 128 + nlo)] = acc[r];
    } else {                                         // a_ss + final state + pooling
#pragma unroll
      for (int r = 0; r < 8; ++r) {
        const int m = r + 8 * half;
        const int n = n0 - 256 + nlo;
        const int node = row0 + m;
        if (node < N_NODES) {
          const float s = state[(size_t)node * DIM + n] + acc[r]
                        + UVl[m * DIM + n] * asv[m * DIM + n];
          state[(size_t)node * DIM + n] = s;
          atomicAdd(graph_acc + (size_t)(node_graph_index[node] - g0) * DIM + n, s);
        }
      }
    }
  }
}

// ---- readout: out = silu(graph_state @ out_W1 + b1) @ out_W2 + b2 --------
__global__ __launch_bounds__(32) void readout_kernel(
    const float* __restrict__ graph_acc,
    const bf16* __restrict__ outW1p, const float* __restrict__ out_b1,
    const float* __restrict__ out_W2, const float* __restrict__ out_b2,
    float* __restrict__ out) {
  __shared__ __align__(16) bf16  gA[16 * DIM];
  __shared__ float Hf[16 * DIM];
  const int lane = threadIdx.x, half = lane >> 4, nlo = lane & 15;
  const int row0 = blockIdx.x * 16;
  {
    const int r = lane >> 1, c0 = (lane & 1) * 64;
    stage_row_bf16(gA + r * DIM + c0, graph_acc + (size_t)(row0 + r) * DIM + c0);
  }
  __syncthreads();
  for (int nt = 0; nt < 8; ++nt) {
    v8f acc; const float b = out_b1[nt * 16 + nlo];
#pragma unroll
    for (int r = 0; r < 8; ++r) acc[r] = b;
    for (int kt = 0; kt < 4; ++kt)
      acc = wmma_bf16(fragA_lds(gA, kt * 32, lane), fragB_p(outW1p, 8, kt, nt, lane), acc);
#pragma unroll
    for (int r = 0; r < 8; ++r)
      Hf[(r + 8 * half) * DIM + nt * 16 + nlo] = silu_f(acc[r]);
  }
  __syncthreads();
  if (lane < 16) {
    float s = out_b2[0];
    for (int c = 0; c < DIM; ++c) s += Hf[lane * DIM + c] * out_W2[c];
    out[row0 + lane] = s;
  }
}

// ---- host side -----------------------------------------------------------

extern "C" void kernel_launch(void* const* d_in, const int* in_sizes, int n_in,
                              void* d_out, int out_size, void* d_ws, size_t ws_size,
                              hipStream_t stream) {
  const float* edge_vec  = (const float*)d_in[0];
  const float* embedding = (const float*)d_in[1];
  const float* phi_W1    = (const float*)d_in[2];
  const float* phi_b1    = (const float*)d_in[3];
  const float* phi_W2    = (const float*)d_in[4];
  const float* phi_b2    = (const float*)d_in[5];
  const float* filt_W    = (const float*)d_in[6];
  const float* filt_b    = (const float*)d_in[7];
  const float* upd_W1    = (const float*)d_in[8];
  const float* upd_b1    = (const float*)d_in[9];
  const float* upd_W2    = (const float*)d_in[10];
  const float* upd_b2    = (const float*)d_in[11];
  const float* out_W1    = (const float*)d_in[12];
  const float* out_b1    = (const float*)d_in[13];
  const float* out_W2    = (const float*)d_in[14];
  const float* out_b2    = (const float*)d_in[15];
  const float* U_W       = (const float*)d_in[16];
  const float* V_W       = (const float*)d_in[17];
  const int* atom_types       = (const int*)d_in[18];
  const int* node_from        = (const int*)d_in[19];
  const int* node_to          = (const int*)d_in[20];
  const int* node_graph_index = (const int*)d_in[21];

  char* ws = (char*)d_ws;
  size_t off = 0;
  auto take = [&](size_t bytes) -> char* {
    char* p = ws + off;
    off = (off + bytes + 255) & ~(size_t)255;
    return p;
  };
  float* state0    = (float*)take((size_t)N_NODES * DIM * 4);
  float* state_msg = (float*)take((size_t)N_NODES * DIM * 4);
  float* state_vec = (float*)take((size_t)N_NODES * 384 * 4);
  float* graph_acc = (float*)take((size_t)N_GRAPHS * DIM * 4);
  bf16* phiW1p = (bf16*)take(128 * 128 * 2);
  bf16* phiW2p = (bf16*)take(128 * 384 * 2);
  bf16* filtWp = (bf16*)take(32 * 384 * 2);
  bf16* updW1p = (bf16*)take(256 * 128 * 2);
  bf16* updW2p = (bf16*)take(128 * 384 * 2);
  bf16* UWp    = (bf16*)take(128 * 128 * 2);
  bf16* VWp    = (bf16*)take(128 * 128 * 2);
  bf16* outW1p = (bf16*)take(128 * 128 * 2);

  auto pack = [&](const float* s, bf16* d, int K, int Ksrc, int N) {
    pack_B_kernel<<<(K * N + 255) / 256, 256, 0, stream>>>(s, d, K, Ksrc, N);
  };
  pack(phi_W1, phiW1p, 128, 128, 128);
  pack(phi_W2, phiW2p, 128, 128, 384);
  pack(filt_W, filtWp, 32, 20, 384);      // K padded 20 -> 32
  pack(upd_W1, updW1p, 256, 256, 128);
  pack(upd_W2, updW2p, 128, 128, 384);
  pack(U_W,    UWp,    128, 128, 128);
  pack(V_W,    VWp,    128, 128, 128);
  pack(out_W1, outW1p, 128, 128, 128);

  gather_embed_kernel<<<N_NODES, DIM, 0, stream>>>(embedding, atom_types, state0, state_msg);
  hipMemsetAsync(state_vec, 0, (size_t)N_NODES * 384 * 4, stream);
  hipMemsetAsync(graph_acc, 0, (size_t)N_GRAPHS * DIM * 4, stream);

  edge_msg_kernel<<<N_EDGES / 16, 32, 0, stream>>>(
      edge_vec, state0, phiW1p, phi_b1, phiW2p, phi_b2, filtWp, filt_b,
      node_from, node_to, state_msg, state_vec);

  node_update_kernel<<<(N_NODES + 15) / 16, 32, 0, stream>>>(
      state_msg, state_vec, UWp, VWp, updW1p, upd_b1, updW2p, upd_b2,
      node_graph_index, graph_acc);

  readout_kernel<<<N_GRAPHS / 16, 32, 0, stream>>>(
      graph_acc, outW1p, out_b1, out_W2, out_b2, (float*)d_out);
}
